// AttentionDecoder_6219112645433
// MI455X (gfx1250) — matrix operand order repaired and checked by reading, not compiled, not run on hardware
//
#include <hip/hip_runtime.h>
#include <hip/hip_bf16.h>

// ---------------------------------------------------------------------------
// AttentionDecoder for MI455X (gfx1250, wave32, WMMA + async-LDS staging)
//   B=64, T_DEC=128, T_ENC=256, D_IN=256, D_ENC=512, H=512
// ---------------------------------------------------------------------------

typedef __attribute__((ext_vector_type(16))) __bf16 v16bf;
typedef __attribute__((ext_vector_type(8)))  __bf16 v8bf;
typedef __attribute__((ext_vector_type(2)))  __bf16 v2bf;
typedef __attribute__((ext_vector_type(8)))  float  v8f;
typedef __attribute__((ext_vector_type(4)))  int    v4i;

typedef __attribute__((address_space(1))) v4i* gptr_v4i;   // global
typedef __attribute__((address_space(3))) v4i* lptr_v4i;   // LDS

#define B_SZ   64
#define T_DECx 128
#define T_ENCx 256
#define D_INx  256
#define D_ENCx 512
#define H_SZ   512
#define G3     1536      // 3*H
#define NBLOCK 64
#define NTHR   256

// ---- CDNA5 async global->LDS copy (ASYNCcnt-tracked DMA) ------------------
#if __has_builtin(__builtin_amdgcn_global_load_async_to_lds_b128)
#define HAVE_ASYNC_LDS 1
#else
#define HAVE_ASYNC_LDS 0
#endif

static __device__ __forceinline__ void async_copy16(const __bf16* gsrc, __bf16* ldst) {
#if HAVE_ASYNC_LDS
  __builtin_amdgcn_global_load_async_to_lds_b128(
      (gptr_v4i)(gsrc), (lptr_v4i)(ldst), 0, 0);
#else
  *(v8bf*)ldst = *(const v8bf*)gsrc;
#endif
}
static __device__ __forceinline__ void async_wait0() {
#if HAVE_ASYNC_LDS
#if __has_builtin(__builtin_amdgcn_s_wait_asynccnt)
  __builtin_amdgcn_s_wait_asynccnt(0);
#else
  asm volatile("s_wait_asynccnt 0" ::: "memory");
#endif
#endif
}

// ---- WMMA fragment loaders (bf16, wave32 layouts) -------------------------
// A 16x32 tile: lanes 0-15 row M=lane, K = {h*16..h*16+15}, h=lane>>4
static __device__ __forceinline__ v16bf load_a_frag(const __bf16* A, int lda,
                                                    int m0, int k0, int lane) {
  int row = m0 + (lane & 15);
  const __bf16* p = A + (size_t)row * lda + k0 + ((lane >> 4) << 4);
  v8bf lo = *(const v8bf*)(p);
  v8bf hi = *(const v8bf*)(p + 8);
  return __builtin_shufflevector(lo, hi, 0,1,2,3,4,5,6,7,8,9,10,11,12,13,14,15);
}
// B 32x16 tile from pre-transposed weights BT (N x K row-major):
// lane = N%16, K contiguous; lane groups split K halves.
static __device__ __forceinline__ v16bf load_b_fragT(const __bf16* BT, int ldb,
                                                     int n0, int k0, int lane) {
  const __bf16* p = BT + (size_t)(n0 + (lane & 15)) * ldb + k0 + ((lane >> 4) << 4);
  return *(const v16bf*)(p);
}

// ---- prep kernels ---------------------------------------------------------
__global__ void f32_to_bf16_kernel(const float* __restrict__ in,
                                   __bf16* __restrict__ out, int n) {
  int i = blockIdx.x * blockDim.x + threadIdx.x;
  if (i < n) out[i] = (__bf16)in[i];
}

// out[n*K+k] = in[(row_off+k)*ld_in + n]   (store W^T slice as bf16)
__global__ void transpose_bf16_kernel(const float* __restrict__ in, int ld_in,
                                      int row_off, __bf16* __restrict__ out,
                                      int N, int K) {
  int i = blockIdx.x * blockDim.x + threadIdx.x;
  if (i < N * K) {
    int n = i / K, k = i - n * K;
    out[(size_t)n * K + k] = (__bf16)in[(size_t)(row_off + k) * ld_in + n];
  }
}

// ---- bulk bf16 WMMA GEMM with async-LDS B staging -------------------------
//  macro-tile: 128 (8 waves x 16) x 64 (4 n-tiles/wave), K-slab 64, dbl-buffer
#define MBLK  128
#define NBLK  64
#define KSLAB 64

__global__ void __launch_bounds__(NTHR)
gemm_lds_kernel(const __bf16* __restrict__ A, const __bf16* __restrict__ BT,
                float* __restrict__ Cf, __bf16* __restrict__ Cbf,
                const float* __restrict__ bias, int M, int N, int K) {
  __shared__ __align__(32) __bf16 sB[2][NBLK * KSLAB];   // 2 x 8 KB
  const int tid  = threadIdx.x;
  const int lane = tid & 31;
  const int wave = tid >> 5;
  const int nblk = N / NBLK;
  const int bm = blockIdx.x / nblk;
  const int bn = blockIdx.x - bm * nblk;
  const int m0    = bm * MBLK + wave * 16;
  const int nbase = bn * NBLK;

  auto stage = [&](int buf, int ks) {
    // 64 rows x 128 B = 8 KB; 256 threads x 2 x 16 B async chunks
#pragma unroll
    for (int j = 0; j < 2; ++j) {
      int c   = tid + j * NTHR;        // 0..511
      int row = c >> 3;                // 0..63
      int cb  = (c & 7) << 3;          // bf16 col offset 0..56
      async_copy16(BT + (size_t)(nbase + row) * K + ks + cb,
                   &sB[buf][row * KSLAB + cb]);
    }
  };

  v8f acc[4] = {};
  stage(0, 0);
  async_wait0();
  __syncthreads();

  const int lrow = (lane & 15) * KSLAB + ((lane >> 4) << 4);   // per-lane LDS base
  for (int ks = 0; ks < K; ks += KSLAB) {
    const int cur = (ks / KSLAB) & 1;
    if (ks + KSLAB < K) stage(cur ^ 1, ks + KSLAB);   // overlap DMA w/ WMMA
#pragma unroll
    for (int kk = 0; kk < KSLAB; kk += 32) {
      v16bf a = load_a_frag(A, K, m0, ks + kk, lane);
      const __bf16* pb = &sB[cur][lrow + kk];
      // distinct fragment registers -> 8 ds_loads clause, partial dscnt waits
      v16bf b0 = *(const v16bf*)(pb + 0 * 16 * KSLAB);
      v16bf b1 = *(const v16bf*)(pb + 1 * 16 * KSLAB);
      v16bf b2 = *(const v16bf*)(pb + 2 * 16 * KSLAB);
      v16bf b3 = *(const v16bf*)(pb + 3 * 16 * KSLAB);
      acc[0] = __builtin_amdgcn_wmma_f32_16x16x32_bf16(false, a, false, b0,
                                                       (short)0, acc[0], false, false);
      acc[1] = __builtin_amdgcn_wmma_f32_16x16x32_bf16(false, a, false, b1,
                                                       (short)0, acc[1], false, false);
      acc[2] = __builtin_amdgcn_wmma_f32_16x16x32_bf16(false, a, false, b2,
                                                       (short)0, acc[2], false, false);
      acc[3] = __builtin_amdgcn_wmma_f32_16x16x32_bf16(false, a, false, b3,
                                                       (short)0, acc[3], false, false);
    }
    __syncthreads();    // everyone done reading `cur`
    async_wait0();      // my staged chunks for `nxt` have landed
    __syncthreads();    // all chunks landed
  }

  const int colb = lane & 15;
  const int rb   = m0 + ((lane >> 4) << 3);
#pragma unroll
  for (int nt = 0; nt < 4; ++nt) {
    int col = nbase + nt * 16 + colb;
    float bv = bias ? bias[col] : 0.f;
#pragma unroll
    for (int i = 0; i < 8; ++i) {
      float v = acc[nt][i] + bv;
      if (Cf)  Cf[(size_t)(rb + i) * N + col]  = v;
      if (Cbf) Cbf[(size_t)(rb + i) * N + col] = (__bf16)v;
    }
  }
}

// ---- device-scope grid barrier -------------------------------------------
static __device__ __forceinline__ void grid_sync(unsigned* counter, unsigned* gen) {
  __syncthreads();
  if (threadIdx.x == 0) {
    __threadfence();
    unsigned g = __hip_atomic_load(gen, __ATOMIC_RELAXED, __HIP_MEMORY_SCOPE_AGENT);
    unsigned arrived = __hip_atomic_fetch_add(counter, 1u, __ATOMIC_ACQ_REL,
                                              __HIP_MEMORY_SCOPE_AGENT);
    if (arrived == NBLOCK - 1) {
      __hip_atomic_store(counter, 0u, __ATOMIC_RELAXED, __HIP_MEMORY_SCOPE_AGENT);
      __hip_atomic_fetch_add(gen, 1u, __ATOMIC_ACQ_REL, __HIP_MEMORY_SCOPE_AGENT);
    } else {
      while (__hip_atomic_load(gen, __ATOMIC_ACQUIRE,
                               __HIP_MEMORY_SCOPE_AGENT) == g) {
        __builtin_amdgcn_s_sleep(1);
      }
    }
    __threadfence();
  }
  __syncthreads();
}

// ---- persistent sequential decoder ---------------------------------------
__global__ void __launch_bounds__(NTHR)
decoder_kernel(const float* __restrict__ att_w2,
               const __bf16* __restrict__ ctxproj_bf,   // (B,Te,De) bf16
               const __bf16* __restrict__ context_bf,   // (B,Te,De) bf16
               const float* __restrict__ gxx,           // (B,T,3H) f32 (bias folded)
               const __bf16* __restrict__ W1hT,         // (De x H) = att_w1[:H]^T
               const __bf16* __restrict__ WxcT,         // (3H x De) = Wx[Din:]^T
               const __bf16* __restrict__ WhT,          // (3H x H)  = Wh^T
               float* __restrict__ hbuf, __bf16* __restrict__ hbf,
               float* __restrict__ hproj,
               float* __restrict__ ctxf, __bf16* __restrict__ ctxbf,
               float* __restrict__ gxc, float* __restrict__ ghh,
               unsigned* __restrict__ bar,
               float* __restrict__ out) {
  const int tid   = threadIdx.x;
  const int lane  = tid & 31;
  const int wave  = tid >> 5;
  const int gwave = blockIdx.x * (NTHR >> 5) + wave;
  unsigned* counter = bar;
  unsigned* gen     = bar + 1;

  __shared__ float s_e[T_ENCx];
  __shared__ float s_tmp[NTHR];

  // h0 = 0 (fresh every launch -> deterministic)
  for (int i = blockIdx.x * NTHR + tid; i < B_SZ * H_SZ; i += NBLOCK * NTHR) {
    hbuf[i] = 0.f; hbf[i] = (__bf16)0.f;
  }
  grid_sync(counter, gen);

  for (int t = 0; t < T_DECx; ++t) {
    // ---- Phase A: hproj = h @ att_w1[:H]  (64x512 @ 512x512) -> 128 tiles
    if (gwave < 128) {
      int mt = gwave >> 5, nt = gwave & 31;
      int m0 = mt << 4, n0 = nt << 4;
      v8f c = {};
      for (int k0 = 0; k0 < H_SZ; k0 += 32) {
        v16bf a = load_a_frag(hbf, H_SZ, m0, k0, lane);
        v16bf b = load_b_fragT(W1hT, H_SZ, n0, k0, lane);
        c = __builtin_amdgcn_wmma_f32_16x16x32_bf16(false, a, false, b,
                                                    (short)0, c, false, false);
      }
      int col = n0 + (lane & 15);
      int rb  = m0 + ((lane >> 4) << 3);
#pragma unroll
      for (int i = 0; i < 8; ++i) hproj[(rb + i) * D_ENCx + col] = c[i];
    }
    grid_sync(counter, gen);

    // ---- Phase B: energies -> softmax -> ctx (one workgroup per batch b)
    {
      const int b = blockIdx.x;
      // wave-invariant operands cached in registers (16 f32 each per lane)
      const float* hpb = hproj + b * D_ENCx;
      float hp[16], w2r[16];
#pragma unroll
      for (int j = 0; j < 16; ++j) {
        hp[j]  = hpb[(j << 5) + lane];
        w2r[j] = att_w2[(j << 5) + lane];
      }
      for (int te = wave; te < T_ENCx; te += (NTHR >> 5)) {
        const __bf16* row = ctxproj_bf + ((size_t)(b * T_ENCx + te)) * D_ENCx;
        __builtin_prefetch(row + 8 * D_ENCx, 0, 3);   // global_prefetch next row
        float s = 0.f;
#pragma unroll
        for (int j = 0; j < 16; ++j) {
          float v = (float)row[(j << 5) + lane] + hp[j];
          s += fmaxf(v, 0.f) * w2r[j];
        }
#pragma unroll
        for (int off = 16; off > 0; off >>= 1) s += __shfl_xor(s, off, 32);
        if (lane == 0) s_e[te] = s;
      }
      __syncthreads();
      // softmax over 256 energies
      float v = s_e[tid];
      s_tmp[tid] = v;
      __syncthreads();
      for (int off = NTHR >> 1; off > 0; off >>= 1) {
        if (tid < off) s_tmp[tid] = fmaxf(s_tmp[tid], s_tmp[tid + off]);
        __syncthreads();
      }
      float m = s_tmp[0];
      __syncthreads();
      float ev = __expf(v - m);
      s_tmp[tid] = ev;
      __syncthreads();
      for (int off = NTHR >> 1; off > 0; off >>= 1) {
        if (tid < off) s_tmp[tid] += s_tmp[tid + off];
        __syncthreads();
      }
      float inv = 1.f / s_tmp[0];
      __syncthreads();
      s_e[tid] = ev * inv;           // attention weights
      __syncthreads();
      // ctx[b,:] = att @ context[b]  (paired bf16 loads: 1 dword/lane/te)
      {
        int d = tid * 2;             // 512 threads-worth covered by 256 x 2
        const __bf16* cb = context_bf + (size_t)b * T_ENCx * D_ENCx + d;
        float acc0 = 0.f, acc1 = 0.f;
#pragma unroll 4
        for (int te = 0; te < T_ENCx; ++te) {
          v2bf pr = *(const v2bf*)(cb + (size_t)te * D_ENCx);
          float w = s_e[te];
          acc0 += w * (float)pr.x;
          acc1 += w * (float)pr.y;
        }
        ctxf[b * D_ENCx + d]      = acc0;
        ctxf[b * D_ENCx + d + 1]  = acc1;
        v2bf pk; pk.x = (__bf16)acc0; pk.y = (__bf16)acc1;
        *(v2bf*)(ctxbf + b * D_ENCx + d) = pk;
      }
    }
    grid_sync(counter, gen);

    // ---- Phase C: gxc = ctx@Wxc + gxx[t], ghh = h@Wh   (64x1536, 384 tiles)
    if (gwave < 384) {
      int mt = gwave / 96, nt = gwave - mt * 96;
      int m0 = mt << 4, n0 = nt << 4;
      v8f cg = {}, ch = {};
      for (int k0 = 0; k0 < H_SZ; k0 += 32) {
        v16bf ac = load_a_frag(ctxbf, D_ENCx, m0, k0, lane);
        v16bf ah = load_a_frag(hbf,   H_SZ,   m0, k0, lane);
        v16bf bc = load_b_fragT(WxcT, D_ENCx, n0, k0, lane);
        v16bf bh = load_b_fragT(WhT,  H_SZ,   n0, k0, lane);
        cg = __builtin_amdgcn_wmma_f32_16x16x32_bf16(false, ac, false, bc,
                                                     (short)0, cg, false, false);
        ch = __builtin_amdgcn_wmma_f32_16x16x32_bf16(false, ah, false, bh,
                                                     (short)0, ch, false, false);
      }
      int col = n0 + (lane & 15);
      int rb  = m0 + ((lane >> 4) << 3);
#pragma unroll
      for (int i = 0; i < 8; ++i) {
        int b = rb + i;
        gxc[b * G3 + col] = cg[i] + gxx[((size_t)b * T_DECx + t) * G3 + col];
        ghh[b * G3 + col] = ch[i];
      }
    }
    grid_sync(counter, gen);

    // ---- Phase D: GRU pointwise update
    for (int i = blockIdx.x * NTHR + tid; i < B_SZ * H_SZ; i += NBLOCK * NTHR) {
      int b = i >> 9, j = i & (H_SZ - 1);
      float xz = gxc[b * G3 + j]            + ghh[b * G3 + j];
      float xr = gxc[b * G3 + H_SZ + j]     + ghh[b * G3 + H_SZ + j];
      float z  = 1.f / (1.f + __expf(-xz));
      float r  = 1.f / (1.f + __expf(-xr));
      float hh = tanhf(gxc[b * G3 + 2 * H_SZ + j] + r * ghh[b * G3 + 2 * H_SZ + j]);
      float ho = hbuf[i];
      float hn = z * ho + (1.f - z) * hh;
      hbuf[i] = hn;
      hbf[i]  = (__bf16)hn;
      out[((size_t)b * T_DECx + t) * H_SZ + j] = hn;
    }
    grid_sync(counter, gen);
  }
}

// ---- host side ------------------------------------------------------------
extern "C" void kernel_launch(void* const* d_in, const int* in_sizes, int n_in,
                              void* d_out, int out_size, void* d_ws, size_t ws_size,
                              hipStream_t stream) {
  (void)in_sizes; (void)n_in; (void)out_size; (void)ws_size;
  const float* inputs  = (const float*)d_in[0];  // (B,T,Din)
  const float* context = (const float*)d_in[1];  // (B,Te,De)
  const float* att_w1  = (const float*)d_in[2];  // (H+De, De)
  const float* att_w2  = (const float*)d_in[3];  // (De,1)
  const float* Wx      = (const float*)d_in[4];  // (Din+De, 3H)
  const float* Wh      = (const float*)d_in[5];  // (H, 3H)
  const float* bias    = (const float*)d_in[6];  // (3H,)
  float* out = (float*)d_out;

  char* ws = (char*)d_ws;
  size_t off = 0;
  auto alloc = [&](size_t bytes) -> void* {
    off = (off + 255) & ~(size_t)255;
    void* p = ws + off;
    off += bytes;
    return p;
  };

  unsigned* bar        = (unsigned*)alloc(256);
  __bf16*   W1hT       = (__bf16*)alloc((size_t)512 * 512 * 2);
  __bf16*   W1cT       = (__bf16*)alloc((size_t)512 * 512 * 2);
  __bf16*   WxxT       = (__bf16*)alloc((size_t)1536 * 256 * 2);
  __bf16*   WxcT       = (__bf16*)alloc((size_t)1536 * 512 * 2);
  __bf16*   WhT        = (__bf16*)alloc((size_t)1536 * 512 * 2);
  __bf16*   inputs_bf  = (__bf16*)alloc((size_t)B_SZ * T_DECx * D_INx * 2);
  __bf16*   context_bf = (__bf16*)alloc((size_t)B_SZ * T_ENCx * D_ENCx * 2);
  __bf16*   ctxproj_bf = (__bf16*)alloc((size_t)B_SZ * T_ENCx * D_ENCx * 2);
  float*    gxx        = (float*)alloc((size_t)B_SZ * T_DECx * G3 * 4);
  float*    hbuf       = (float*)alloc((size_t)B_SZ * H_SZ * 4);
  __bf16*   hbf        = (__bf16*)alloc((size_t)B_SZ * H_SZ * 2);
  float*    hproj      = (float*)alloc((size_t)B_SZ * D_ENCx * 4);
  float*    ctxf       = (float*)alloc((size_t)B_SZ * D_ENCx * 4);
  __bf16*   ctxbf      = (__bf16*)alloc((size_t)B_SZ * D_ENCx * 2);
  float*    gxc        = (float*)alloc((size_t)B_SZ * G3 * 4);
  float*    ghh        = (float*)alloc((size_t)B_SZ * G3 * 4);

  (void)hipMemsetAsync(bar, 0, 256, stream);

  const int thr = 256;
  // W^T slices (bf16)
  transpose_bf16_kernel<<<(512 * 512 + thr - 1) / thr, thr, 0, stream>>>(
      att_w1, D_ENCx, 0, W1hT, 512, 512);
  transpose_bf16_kernel<<<(512 * 512 + thr - 1) / thr, thr, 0, stream>>>(
      att_w1, D_ENCx, H_SZ, W1cT, 512, 512);
  transpose_bf16_kernel<<<(1536 * 256 + thr - 1) / thr, thr, 0, stream>>>(
      Wx, G3, 0, WxxT, 1536, 256);
  transpose_bf16_kernel<<<(1536 * 512 + thr - 1) / thr, thr, 0, stream>>>(
      Wx, G3, D_INx, WxcT, 1536, 512);
  transpose_bf16_kernel<<<(1536 * 512 + thr - 1) / thr, thr, 0, stream>>>(
      Wh, G3, 0, WhT, 1536, 512);
  // bf16 activations
  {
    int n = B_SZ * T_DECx * D_INx;
    f32_to_bf16_kernel<<<(n + thr - 1) / thr, thr, 0, stream>>>(inputs, inputs_bf, n);
  }
  {
    int n = B_SZ * T_ENCx * D_ENCx;
    f32_to_bf16_kernel<<<(n + thr - 1) / thr, thr, 0, stream>>>(context, context_bf, n);
  }
  // Hoisted GEMMs (async-LDS staged, double-buffered):
  //   ctxproj = context @ att_w1[H:,:]   (16384 x 512 x 512)
  gemm_lds_kernel<<<(16384 / MBLK) * (512 / NBLK), NTHR, 0, stream>>>(
      context_bf, W1cT, nullptr, ctxproj_bf, nullptr, B_SZ * T_ENCx, D_ENCx, D_ENCx);
  //   gxx = inputs @ Wx[:Din,:] + b      (8192 x 1536 x 256)
  gemm_lds_kernel<<<(8192 / MBLK) * (1536 / NBLK), NTHR, 0, stream>>>(
      inputs_bf, WxxT, gxx, nullptr, bias, B_SZ * T_DECx, G3, D_INx);
  // Sequential persistent decode
  decoder_kernel<<<NBLOCK, NTHR, 0, stream>>>(att_w2, ctxproj_bf, context_bf,
                                              gxx, W1hT, WxcT, WhT,
                                              hbuf, hbf, hproj, ctxf, ctxbf,
                                              gxc, ghh, bar, out);
}
